// G_CLstm_10548439679398
// MI455X (gfx1250) — matrix operand order repaired and checked by reading, not compiled, not run on hardware
//
#include <hip/hip_runtime.h>
#include <hip/hip_bf16.h>
#include <math.h>

typedef __attribute__((ext_vector_type(16))) __bf16 v16bf;
typedef __attribute__((ext_vector_type(8)))  __bf16 v8bf;
typedef __attribute__((ext_vector_type(8)))  float  v8f;

#define GROUPS 2
#define HLEN   160          // STEP
#define FOURH  640
#define BATCH  64
#define TSTEPS 1000
#define CHN    64
#define FEAT   5
#define DMODEL 320          // CHN*FEAT
#define NT     40           // FOURH / 16
#define NKO    5            // HLEN / 32

// Dynamic-LDS partition for the recurrent kernel (bytes)
#define LDS_WHH    204800                       // [40][5][32][16] bf16 swizzled Whh
#define LDS_GATES  40960                        // 16 x 640 f32 gate exchange
#define LDS_H      5120                         // 16 x 160 bf16 hidden state
#define LDS_TOTAL  (LDS_WHH + LDS_GATES + LDS_H)  // 250,880 B < 320 KB WGP LDS

#if defined(__has_builtin)
#if __has_builtin(__builtin_amdgcn_global_load_async_to_lds_b128) && \
    __has_builtin(__builtin_amdgcn_s_wait_asynccnt)
#define USE_ASYNC_LDS 1
#endif
#endif

#ifdef USE_ASYNC_LDS
// Exact parameter types per hipcc diagnostic:
//   param0: 'int __attribute__((vector_size(16))) __device__ *'  (global side)
//   param1: LDS side, same vector type in addrspace(3)
typedef int v4i_vs __attribute__((vector_size(4 * sizeof(int))));
typedef __attribute__((address_space(1))) v4i_vs* as1_v4i;
typedef __attribute__((address_space(3))) v4i_vs* as3_v4i;
#endif

__device__ __forceinline__ float fast_sig(float x) {
    return 1.0f / (1.0f + __expf(-x));
}
__device__ __forceinline__ float fast_tanh(float x) {
    x = fminf(fmaxf(x, -15.0f), 15.0f);
    float e = __expf(2.0f * x);
    return (e - 1.0f) / (e + 1.0f);
}

// ---------------------------------------------------------------------------
// Convert a [G,160,640] f32 weight into the per-lane swizzled bf16 WMMA-B
// layout: dst[(((g*40+nt)*5+ko)*32+lane)*16+j] holds
// W[K = ko*32+(lane>>4)*8+(j&7)+(j>>3)*16, N = nt*16+(lane&15)], so each GEMM lane does one
// contiguous 32B load per (nt,ko) tile.
// ---------------------------------------------------------------------------
__global__ void k_convert_w(const float* __restrict__ src, __bf16* __restrict__ dst) {
    int e = blockIdx.x * 256 + threadIdx.x;          // 204800 elements total
    int g    = e / 102400;
    int rem  = e % 102400;
    int nt   = rem / 2560;
    int rem2 = rem % 2560;
    int ko   = rem2 / 512;
    int rem3 = rem2 % 512;
    int lane = rem3 >> 4;
    int j    = rem3 & 15;
    int K = ko * 32 + ((lane >> 4) << 3) + (j & 7) + ((j >> 3) << 4);
    int N = nt * 16 + (lane & 15);
    dst[e] = (__bf16)src[(g * HLEN + K) * FOURH + N];
}

// ---------------------------------------------------------------------------
// input [B,C,T,F] f32  ->  x1 [B,T,D=320] bf16  (d = c*F + f)
// ---------------------------------------------------------------------------
__global__ void k_reorder_in(const float* __restrict__ in, __bf16* __restrict__ x1) {
    int idx = blockIdx.x * 256 + threadIdx.x;        // 20,480,000 total
    int d = idx % DMODEL;
    int t = (idx / DMODEL) % TSTEPS;
    int b = idx / (DMODEL * TSTEPS);
    int c = d / FEAT, f = d % FEAT;
    x1[idx] = (__bf16)in[(((size_t)b * CHN + c) * TSTEPS + t) * FEAT + f];
}

// ---------------------------------------------------------------------------
// Parallel ih-GEMM: xg[m,g,o] = bias[g,o] + sum_k x[m, g*H + k] * Wih[g,k,o]
// m = b*T + t over 64000 rows; one wave per 16x16 tile, 5 K-steps of
// v_wmma_f32_16x16x32_bf16; 8 waves/block share the M-tile (A stays in L0/L2).
// ---------------------------------------------------------------------------
__global__ void __launch_bounds__(256)
k_xgates(const __bf16* __restrict__ x, const __bf16* __restrict__ wsw,
         const float* __restrict__ bias, float* __restrict__ xg) {
    int tid = threadIdx.x, w = tid >> 5, lane = tid & 31;
    int mt  = blockIdx.x / 10;                       // 0..3999
    int rem = blockIdx.x % 10;
    int g   = rem / 5;
    int nt  = (rem % 5) * 8 + w;                     // 0..39
    int m0  = mt * 16;

    const __bf16* xrow  = x + (size_t)(m0 + (lane & 15)) * DMODEL + g * HLEN;
    const __bf16* wbase = wsw + (size_t)(g * NT + nt) * NKO * 512;

    v8f acc;
#pragma unroll
    for (int j = 0; j < 8; ++j) acc[j] = 0.0f;

#pragma unroll
    for (int ko = 0; ko < NKO; ++ko) {
        int kb = ((lane >> 4) << 3) + ko * 32;       // ISA 16-bit A layout
        v8bf lo = *(const v8bf*)(xrow + kb);
        v8bf hi = *(const v8bf*)(xrow + kb + 16);
        v16bf a;
#pragma unroll
        for (int j = 0; j < 8; ++j) { a[j] = lo[j]; a[j + 8] = hi[j]; }
        v16bf bm = *(const v16bf*)(wbase + (size_t)ko * 512 + lane * 16);
        acc = __builtin_amdgcn_wmma_f32_16x16x32_bf16(
                  false, a, false, bm, (short)0, acc, false, false);
    }

    int o = nt * 16 + (lane & 15);
    float bv = bias[g * FOURH + o];
#pragma unroll
    for (int v = 0; v < 8; ++v) {
        int mrow = ((lane >> 4) << 3) + v;           // C/D layout
        xg[((size_t)(m0 + mrow) * GROUPS + g) * FOURH + o] = acc[v] + bv;
    }
}

// ---------------------------------------------------------------------------
// Persistent recurrence: 8 WGs = 4 batch-tiles x 2 groups; each WG owns an
// independent 16-row LSTM recurrence over all 1000 steps.
// Per step:
//   - x-gates for this t staged into registers FIRST (HBM latency hides under
//     the WMMA chain), prefetch warms t+2.
//   - gates = h @ Whh: 200 bf16 WMMAs across 8 waves; Whh lives in LDS
//     (copied once at start via async-to-LDS when available).
//   - elementwise update with c in registers; h -> LDS (bf16) for next step.
// STAGE 1 scatters h through the snd_index permutation (bf16 feed to stage 2);
// STAGE 2 writes the final f32 output with the [B,T,C,F]->[B,C,T,F] transpose.
// ---------------------------------------------------------------------------
template <int STAGE>
__global__ void __launch_bounds__(256)
k_lstm(const float* __restrict__ xg, const __bf16* __restrict__ whh_sw,
       __bf16* __restrict__ out_bf16, float* __restrict__ out_f32) {
    extern __shared__ __align__(16) char smem[];
    __bf16* whh_lds   = (__bf16*)smem;                         // 204,800 B
    float*  gates_lds = (float*)(smem + LDS_WHH);              //  40,960 B
    __bf16* h_lds     = (__bf16*)(smem + LDS_WHH + LDS_GATES); //   5,120 B

    int tid = threadIdx.x, w = tid >> 5, lane = tid & 31;
    int bt = blockIdx.x >> 1;                                // batch tile 0..3
    int g  = blockIdx.x & 1;

    // fixed thread <-> (row, k-slab) mapping for the elementwise phase
    int er  = tid >> 4;                                      // 0..15
    int ek0 = (tid & 15) * 10;                               // 0..150
    int eb  = bt * 16 + er;                                  // batch index

    // ---- one-time: Whh (this group) -> LDS, 12800 x 16B chunks -------------
    const __bf16* wb0 = whh_sw + (size_t)g * NT * NKO * 512;
#ifdef USE_ASYNC_LDS
    for (int i = tid; i < 12800; i += 256) {
        __builtin_amdgcn_global_load_async_to_lds_b128(
            (as1_v4i)(wb0 + (size_t)i * 8),
            (as3_v4i)(whh_lds + (size_t)i * 8), 0, 0);
    }
    __builtin_amdgcn_s_wait_asynccnt(0);
#else
    for (int i = tid; i < 12800; i += 256) {
        *(v8bf*)(whh_lds + (size_t)i * 8) = *(const v8bf*)(wb0 + (size_t)i * 8);
    }
#endif

    float c[10];
#pragma unroll
    for (int u = 0; u < 10; ++u) {
        c[u] = 0.0f;
        h_lds[er * HLEN + ek0 + u] = (__bf16)0.0f;
    }
    __syncthreads();

    for (int t = 0; t < TSTEPS; ++t) {
        // ---- stage x-gates for this step into registers (overlaps GEMM) ----
        const float* xgrow = xg + ((size_t)(eb * TSTEPS + t) * GROUPS + g) * FOURH + ek0;
        float xr[40];
#pragma unroll
        for (int q = 0; q < 4; ++q)
#pragma unroll
            for (int u = 0; u < 10; ++u)
                xr[q * 10 + u] = xgrow[q * HLEN + u];
        if (t + 2 < TSTEPS) {       // warm L2/L0 for t+2
            const float* pf = xgrow + 2 * (size_t)GROUPS * FOURH;
            __builtin_prefetch(pf, 0, 0);
            __builtin_prefetch(pf + 2 * HLEN, 0, 0);
        }

        // ---- GEMM phase: gates_lds = h @ Whh (Whh from LDS) ----------------
        v8f acc[5];
#pragma unroll
        for (int s = 0; s < 5; ++s)
#pragma unroll
            for (int j = 0; j < 8; ++j) acc[s][j] = 0.0f;

#pragma unroll
        for (int ko = 0; ko < NKO; ++ko) {
            int r  = lane & 15;
            int kb = ((lane >> 4) << 3) + ko * 32;
            v8bf lo = *(const v8bf*)&h_lds[r * HLEN + kb];
            v8bf hi = *(const v8bf*)&h_lds[r * HLEN + kb + 16];
            v16bf a;
#pragma unroll
            for (int j = 0; j < 8; ++j) { a[j] = lo[j]; a[j + 8] = hi[j]; }
#pragma unroll
            for (int s = 0; s < 5; ++s) {
                int nt = w + 8 * s;
                v16bf bm = *(const v16bf*)(whh_lds + ((size_t)nt * NKO + ko) * 512 + lane * 16);
                acc[s] = __builtin_amdgcn_wmma_f32_16x16x32_bf16(
                             false, a, false, bm, (short)0, acc[s], false, false);
            }
        }
#pragma unroll
        for (int s = 0; s < 5; ++s) {
            int col = (w + 8 * s) * 16 + (lane & 15);
#pragma unroll
            for (int v = 0; v < 8; ++v) {
                int row = ((lane >> 4) << 3) + v;
                gates_lds[row * FOURH + col] = acc[s][v];
            }
        }
        __syncthreads();

        // ---- elementwise phase --------------------------------------------
#pragma unroll
        for (int u = 0; u < 10; ++u) {
            int k = ek0 + u;
            float iv = gates_lds[er * FOURH + k]            + xr[u];
            float fv = gates_lds[er * FOURH + HLEN + k]     + xr[10 + u];
            float gv = gates_lds[er * FOURH + 2 * HLEN + k] + xr[20 + u];
            float ov = gates_lds[er * FOURH + 3 * HLEN + k] + xr[30 + u];
            float cn = fast_sig(fv) * c[u] + fast_sig(iv) * fast_tanh(gv);
            c[u] = cn;
            float h = fast_sig(ov) * fast_tanh(cn);
            h_lds[er * HLEN + k] = (__bf16)h;
            if (STAGE == 1) {
                // snd_index permutation fused at the producer:
                // source d = g*160 + k -> dest p = (k/5)*10 + g*5 + (k%5)
                int p = (k / FEAT) * 10 + g * FEAT + (k % FEAT);
                out_bf16[(size_t)(eb * TSTEPS + t) * DMODEL + p] = (__bf16)h;
            } else {
                // final transpose [B,T,C,F] -> [B,C,T,F]
                int d  = g * HLEN + k;
                int cc = d / FEAT, f = d % FEAT;
                out_f32[(((size_t)eb * CHN + cc) * TSTEPS + t) * FEAT + f] = h;
            }
        }
        __syncthreads();   // h ready for next step; gates_lds free for reuse
    }
}

// ---------------------------------------------------------------------------
extern "C" void kernel_launch(void* const* d_in, const int* in_sizes, int n_in,
                              void* d_out, int out_size, void* d_ws, size_t ws_size,
                              hipStream_t stream) {
    const float* input = (const float*)d_in[0];
    const float* Wih1  = (const float*)d_in[1];
    const float* Whh1  = (const float*)d_in[2];
    const float* b1    = (const float*)d_in[3];
    const float* Wih2  = (const float*)d_in[4];
    const float* Whh2  = (const float*)d_in[5];
    const float* b2    = (const float*)d_in[6];
    // d_in[7] = snd_index : recomputed analytically on device, not needed

    char* ws = (char*)d_ws;
    const size_t WSW_BYTES = (size_t)GROUPS * HLEN * FOURH * 2;   // 409,600
    const size_t X_BYTES   = (size_t)BATCH * TSTEPS * DMODEL * 2; // 40,960,000
    __bf16* wih1_sw = (__bf16*)(ws);
    __bf16* whh1_sw = (__bf16*)(ws + 1 * WSW_BYTES);
    __bf16* wih2_sw = (__bf16*)(ws + 2 * WSW_BYTES);
    __bf16* whh2_sw = (__bf16*)(ws + 3 * WSW_BYTES);
    __bf16* x1      = (__bf16*)(ws + 4 * WSW_BYTES);
    __bf16* x2      = (__bf16*)(ws + 4 * WSW_BYTES + X_BYTES);
    float*  xg      = (float*) (ws + 4 * WSW_BYTES + 2 * X_BYTES); // 327.7 MB

    // allow >64KB dynamic LDS for the persistent recurrent kernels
    (void)hipFuncSetAttribute((const void*)k_lstm<1>,
        hipFuncAttributeMaxDynamicSharedMemorySize, LDS_TOTAL);
    (void)hipFuncSetAttribute((const void*)k_lstm<2>,
        hipFuncAttributeMaxDynamicSharedMemorySize, LDS_TOTAL);

    // weight swizzle (bf16, WMMA-B lane layout)
    k_convert_w<<<800, 256, 0, stream>>>(Wih1, wih1_sw);
    k_convert_w<<<800, 256, 0, stream>>>(Whh1, whh1_sw);
    k_convert_w<<<800, 256, 0, stream>>>(Wih2, wih2_sw);
    k_convert_w<<<800, 256, 0, stream>>>(Whh2, whh2_sw);
    // [B,C,T,F] -> [B,T,D] bf16
    k_reorder_in<<<80000, 256, 0, stream>>>(input, x1);

    // stage 1: parallel ih-GEMM, then persistent recurrence
    k_xgates<<<40000, 256, 0, stream>>>(x1, wih1_sw, b1, xg);
    k_lstm<1><<<8, 256, LDS_TOTAL, stream>>>(xg, whh1_sw, x2, nullptr);

    // stage 2: same pipeline on the permuted activations, final f32 output
    k_xgates<<<40000, 256, 0, stream>>>(x2, wih2_sw, b2, xg);
    k_lstm<2><<<8, 256, LDS_TOTAL, stream>>>(xg, whh2_sw, nullptr, (float*)d_out);
}
